// MultiHeadAttention_4939212391225
// MI455X (gfx1250) — compile-verified, matrix operand order checked by reading, not run on hardware
//
#include <hip/hip_runtime.h>

// ---------------------------------------------------------------------------
// MI455X (gfx1250) fused multi-head attention forward.
//   1) cvt x -> bf16;  cvt+transpose Wq/Wk/Wv/Wo -> bf16 (N x K)
//   2) WMMA GEMM (32x64 wave tile, ping-pong double-buffered k-loop):
//      Q,K as [b,h,t,d]; V as [b,h,d,t] (transposed -> contiguous PV frags)
//   3) flash-attention (online softmax) with v_wmma_f32_16x16x32_bf16;
//      V loads + next-block K loads hidden under softmax VALU
//   4) WMMA GEMM: ctx @ Wo^T + bias -> f32 out
// ---------------------------------------------------------------------------

typedef __attribute__((ext_vector_type(16))) __bf16 v16bf;
typedef __attribute__((ext_vector_type(8)))  __bf16 v8bf;
typedef __attribute__((ext_vector_type(8)))  float  v8f;

#define B_     4
#define T_     2048
#define H_     16
#define HD_    64
#define D_     1024
#define MROWS  (B_*T_)        // 8192

__device__ __forceinline__ v8f wmma_bf16(v16bf a, v16bf b, v8f c) {
  // D = A(16x32 bf16) * B(32x16 bf16) + C(16x16 f32)
  return __builtin_amdgcn_wmma_f32_16x16x32_bf16(
      /*neg_a=*/false, a, /*neg_b=*/false, b,
      /*c_mod=*/(short)0, c, /*reuse_a=*/false, /*reuse_b=*/false);
}

// Per-lane 16-bit A/B fragment: elements e<8 at p[e], e>=8 at p[16+(e-8)]
// (caller already applied the half*8 K-offset). Two aligned 16-byte loads.
__device__ __forceinline__ v16bf load_frag(const __bf16* p) {
  v8bf lo = *(const v8bf*)(p);
  v8bf hi = *(const v8bf*)(p + 16);
  v16bf r;
#pragma unroll
  for (int i = 0; i < 8; ++i) { r[i] = lo[i]; r[i + 8] = hi[i]; }
  return r;
}

// ---------------------------------------------------------------- converts --
__global__ void cvt_f32_bf16(const float* __restrict__ in,
                             __bf16* __restrict__ out, int n) {
  int i = blockIdx.x * blockDim.x + threadIdx.x;
  int stride = gridDim.x * blockDim.x;
  for (; i < n; i += stride) out[i] = (__bf16)in[i];
}

// in: K x N (row-major f32)  ->  out: N x K (row-major bf16)
__global__ void cvt_transpose_bf16(const float* __restrict__ in,
                                   __bf16* __restrict__ out, int K, int N) {
  int n = blockIdx.x * 16 + threadIdx.x;
  int k = blockIdx.y * 16 + threadIdx.y;
  if (n < N && k < K) out[(size_t)n * K + k] = (__bf16)in[(size_t)k * N + n];
}

// -------------------------------------------------------------- WMMA GEMM --
// C[M,N] = A[M,K] * Bt[N,K]^T ; wave -> 32x64 tile (B frags reused 2x),
// 8 waves/block -> 256x64. k-loop unrolled 2x with ping-pong fragment sets:
// set X computes while set Y loads -> no rotation copies, staggered waits.
// mode 0: f32 out row-major + bias   (final projection)
// mode 1: bf16 out at [b,h,t,d]      (Q, K)
// mode 2: bf16 out at [b,h,d,t]      (V transposed for the PV stage)
#define GEMM_LOAD(A0v, A1v, Bv, koff)                                        \
  do {                                                                       \
    A0v = load_frag(ar0 + (koff));                                           \
    A1v = load_frag(ar1 + (koff));                                           \
    _Pragma("unroll")                                                        \
    for (int nt = 0; nt < 4; ++nt) Bv[nt] = load_frag(br[nt] + (koff));      \
  } while (0)

#define GEMM_MMA(A0v, A1v, Bv)                                               \
  do {                                                                       \
    _Pragma("unroll")                                                        \
    for (int nt = 0; nt < 4; ++nt) acc[nt] = wmma_bf16(A0v, Bv[nt], acc[nt]);\
    _Pragma("unroll")                                                        \
    for (int nt = 0; nt < 4; ++nt)                                           \
      acc[4 + nt] = wmma_bf16(A1v, Bv[nt], acc[4 + nt]);                     \
  } while (0)

__global__ void __launch_bounds__(256)
gemm_wmma_bf16(const __bf16* __restrict__ A, const __bf16* __restrict__ Bt,
               float* __restrict__ outf, __bf16* __restrict__ outb,
               const float* __restrict__ bias, int M, int N, int K, int mode) {
  const int lane = threadIdx.x & 31;
  const int wave = threadIdx.x >> 5;
  const int half = lane >> 4;
  const int ln   = lane & 15;
  const int m0   = blockIdx.y * 256 + wave * 32;
  const int n0   = blockIdx.x * 64;
  if (m0 >= M) return;

  const v8f zero = {0.f, 0.f, 0.f, 0.f, 0.f, 0.f, 0.f, 0.f};
  v8f acc[8];                       // [g*4+nt]: g=0 rows m0+0..15, g=1 +16..31
#pragma unroll
  for (int t = 0; t < 8; ++t) acc[t] = zero;

  const __bf16* ar0 = A + (size_t)(m0 + ln) * K + half * 8;
  const __bf16* ar1 = ar0 + (size_t)16 * K;
  const __bf16* br[4];
#pragma unroll
  for (int nt = 0; nt < 4; ++nt)
    br[nt] = Bt + (size_t)(n0 + nt * 16 + ln) * K + half * 8;

  v16bf xa0, xa1, xb[4];            // set X
  v16bf ya0, ya1, yb[4];            // set Y
  GEMM_LOAD(xa0, xa1, xb, 0);

  int k0 = 0;                       // K is a multiple of 64
  for (; k0 < K - 64; k0 += 64) {
    GEMM_LOAD(ya0, ya1, yb, k0 + 32);   // Y loads fly while X computes
    GEMM_MMA(xa0, xa1, xb);
    GEMM_LOAD(xa0, xa1, xb, k0 + 64);   // X reloads while Y computes
    GEMM_MMA(ya0, ya1, yb);
  }
  GEMM_LOAD(ya0, ya1, yb, K - 32);
  GEMM_MMA(xa0, xa1, xb);
  GEMM_MMA(ya0, ya1, yb);

#pragma unroll
  for (int g = 0; g < 2; ++g) {
#pragma unroll
    for (int nt = 0; nt < 4; ++nt) {
#pragma unroll
      for (int i = 0; i < 8; ++i) {
        const int m = m0 + g * 16 + i + 8 * half;   // C layout: VGPR i, half
        const int n = n0 + nt * 16 + ln;
        const float v = acc[g * 4 + nt][i];
        if (mode == 0) {
          outf[(size_t)m * N + n] = v + bias[n];
        } else {
          const int b = m >> 11, t = m & (T_ - 1);  // T_ = 2048
          const int h = n >> 6,  d = n & (HD_ - 1);
          const size_t idx = (mode == 1)
              ? (((size_t)(b * H_ + h) * T_ + t) * HD_ + d)
              : (((size_t)(b * H_ + h) * HD_ + d) * T_ + t);
          outb[idx] = (__bf16)v;
        }
      }
    }
  }
}

// --------------------------------------------------------- flash attention --
// One wave per (b,h, 16-query block). Online softmax, scores never hit HBM.
__global__ void __launch_bounds__(256)
attn_wmma(const __bf16* __restrict__ Qh,   // [B*H, T, 64]
          const __bf16* __restrict__ Kh,   // [B*H, T, 64]
          const __bf16* __restrict__ Vt,   // [B*H, 64, T]
          __bf16* __restrict__ ctx) {      // [B*T, D]
  __shared__ __bf16 lds_p[8 * 16 * 32];    // per-wave 16x32 P transpose tile
  const int lane = threadIdx.x & 31;
  const int wave = threadIdx.x >> 5;
  const int half = lane >> 4;
  const int ln   = lane & 15;
  const int gw   = blockIdx.x * 8 + wave;
  const int qb   = gw & (T_ / 16 - 1);     // 128 query blocks per (b,h)
  const int bh   = gw / (T_ / 16);
  const int q0   = qb * 16;
  const int h    = bh & (H_ - 1);
  const int b    = bh / H_;
  __bf16* pbuf   = &lds_p[wave * 16 * 32];

  // Q A-fragments (16x64 = two 16x32 K-slices), resident for the whole loop.
  const __bf16* qrow = Qh + ((size_t)bh * T_ + q0 + ln) * HD_ + half * 8;
  const v16bf aq0 = load_frag(qrow + 0);
  const v16bf aq1 = load_frag(qrow + 32);

  const v8f zero = {0.f, 0.f, 0.f, 0.f, 0.f, 0.f, 0.f, 0.f};
  v8f o[4];
#pragma unroll
  for (int nt = 0; nt < 4; ++nt) o[nt] = zero;
  float ms[8], ls[8];
#pragma unroll
  for (int i = 0; i < 8; ++i) { ms[i] = -1e30f; ls[i] = 0.f; }

  const float scale = 0.125f;              // 1/sqrt(64)
  const int kbmax = (q0 + 15) >> 5;        // causal upper bound (32-key blocks)

  const __bf16* krow = Kh + ((size_t)bh * T_ + ln) * HD_ + half * 8;
  // prologue: K fragments for kb = 0 (keys ln / 16+ln, k-slices 0..31/32..63)
  v16bf bk[4];
  bk[0] = load_frag(krow);
  bk[1] = load_frag(krow + 32);
  bk[2] = load_frag(krow + (size_t)16 * HD_);
  bk[3] = load_frag(krow + (size_t)16 * HD_ + 32);

  for (int kb = 0;; ++kb) {
    const int kk0 = kb * 32;
    const bool last = (kb == kbmax);
    // ---- S(16x32) = Q @ K^T : 4 WMMAs on prefetched fragments ------------
    v8f c0 = zero, c1 = zero;
    c0 = wmma_bf16(aq0, bk[0], c0);
    c0 = wmma_bf16(aq1, bk[1], c0);
    c1 = wmma_bf16(aq0, bk[2], c1);
    c1 = wmma_bf16(aq1, bk[3], c1);

    // ---- issue V loads + NEXT block's K loads; softmax hides them --------
    v16bf bv[4];
#pragma unroll
    for (int nt = 0; nt < 4; ++nt) {
      const __bf16* vrow =
          Vt + ((size_t)bh * HD_ + nt * 16 + ln) * T_ + kk0 + half * 8;
      bv[nt] = load_frag(vrow);
    }
    v16bf bkn[4];
    if (!last) {
      const __bf16* krn = krow + (size_t)(kk0 + 32) * HD_;
      bkn[0] = load_frag(krn);
      bkn[1] = load_frag(krn + 32);
      bkn[2] = load_frag(krn + (size_t)16 * HD_);
      bkn[3] = load_frag(krn + (size_t)16 * HD_ + 32);
    }

    const int key0 = kk0 + ln, key1 = key0 + 16;
    // Online softmax; row q lives in one 16-lane half (xor<=8 stays inside).
#pragma unroll
    for (int i = 0; i < 8; ++i) {
      const int q = q0 + i + 8 * half;
      float s0 = c0[i] * scale; if (key0 > q) s0 = -1e30f;
      float s1 = c1[i] * scale; if (key1 > q) s1 = -1e30f;
      float bm = fmaxf(s0, s1);
#pragma unroll
      for (int mk = 1; mk < 16; mk <<= 1) bm = fmaxf(bm, __shfl_xor(bm, mk, 32));
      const float mnew  = fmaxf(ms[i], bm);
      const float alpha = __expf(ms[i] - mnew);
      const float p0 = __expf(s0 - mnew);
      const float p1 = __expf(s1 - mnew);
      float bs = p0 + p1;
#pragma unroll
      for (int mk = 1; mk < 16; mk <<= 1) bs += __shfl_xor(bs, mk, 32);
      ls[i] = ls[i] * alpha + bs;
      ms[i] = mnew;
#pragma unroll
      for (int nt = 0; nt < 4; ++nt) o[nt][i] *= alpha;
      // C-layout -> LDS [row][key], to be re-read in A-layout.
      pbuf[(i + 8 * half) * 32 + ln]      = (__bf16)p0;
      pbuf[(i + 8 * half) * 32 + 16 + ln] = (__bf16)p1;
    }
    asm volatile("s_wait_dscnt 0" ::: "memory");  // per-wave LDS RAW fence

    // P A-fragment (16x32) from LDS: two ds_load_b128 per lane.
    v16bf pf;
    {
      const __bf16* pr = &pbuf[ln * 32 + half * 8];
      v8bf lo = *(const v8bf*)pr;
      v8bf hi = *(const v8bf*)(pr + 16);
#pragma unroll
      for (int i = 0; i < 8; ++i) { pf[i] = lo[i]; pf[i + 8] = hi[i]; }
    }
    // ---- O(16x64) += P(16x32) @ V(32x64) : 4 WMMAs (V already loaded) ----
#pragma unroll
    for (int nt = 0; nt < 4; ++nt) o[nt] = wmma_bf16(pf, bv[nt], o[nt]);

    if (last) break;
#pragma unroll
    for (int nt = 0; nt < 4; ++nt) bk[nt] = bkn[nt];
  }

  // Normalize and emit ctx as bf16 [b, t, h*64+d] for the output GEMM.
#pragma unroll
  for (int i = 0; i < 8; ++i) {
    const float inv = 1.f / ls[i];
    const int t = q0 + i + 8 * half;
#pragma unroll
    for (int nt = 0; nt < 4; ++nt) {
      const int dcol = h * HD_ + nt * 16 + ln;
      ctx[(size_t)(b * T_ + t) * D_ + dcol] = (__bf16)(o[nt][i] * inv);
    }
  }
}

// ------------------------------------------------------------------- host --
extern "C" void kernel_launch(void* const* d_in, const int* in_sizes, int n_in,
                              void* d_out, int out_size, void* d_ws, size_t ws_size,
                              hipStream_t stream) {
  (void)in_sizes; (void)n_in; (void)out_size; (void)ws_size;
  const float* x  = (const float*)d_in[0];
  const float* Wq = (const float*)d_in[1];
  const float* Wk = (const float*)d_in[2];
  const float* Wv = (const float*)d_in[3];
  const float* Wo = (const float*)d_in[4];
  const float* bo = (const float*)d_in[5];
  float* out = (float*)d_out;

  char* ws = (char*)d_ws;
  const size_t szX = (size_t)MROWS * D_ * sizeof(__bf16);   // 16 MiB
  const size_t szW = (size_t)D_ * D_ * sizeof(__bf16);      // 2 MiB
  __bf16* xb  = (__bf16*)(ws);
  __bf16* WqT = (__bf16*)(ws + szX);
  __bf16* WkT = (__bf16*)(ws + szX + szW);
  __bf16* WvT = (__bf16*)(ws + szX + 2 * szW);
  __bf16* WoT = (__bf16*)(ws + szX + 3 * szW);
  __bf16* Qh  = (__bf16*)(ws + szX + 4 * szW);
  __bf16* Kh  = (__bf16*)(ws + 2 * szX + 4 * szW);
  __bf16* Vt  = (__bf16*)(ws + 3 * szX + 4 * szW);
  __bf16* ctx = (__bf16*)(ws + 4 * szX + 4 * szW);          // total 88 MiB

  // 1) precision conversion + weight transposes
  cvt_f32_bf16<<<8192, 256, 0, stream>>>(x, xb, MROWS * D_);
  dim3 tb(16, 16), tg(D_ / 16, D_ / 16);
  cvt_transpose_bf16<<<tg, tb, 0, stream>>>(Wq, WqT, D_, D_);
  cvt_transpose_bf16<<<tg, tb, 0, stream>>>(Wk, WkT, D_, D_);
  cvt_transpose_bf16<<<tg, tb, 0, stream>>>(Wv, WvT, D_, D_);
  cvt_transpose_bf16<<<tg, tb, 0, stream>>>(Wo, WoT, D_, D_);

  // 2) QKV projections (Q,K head layout; V transposed head layout)
  dim3 gg(D_ / 64, MROWS / 256);
  gemm_wmma_bf16<<<gg, 256, 0, stream>>>(xb, WqT, nullptr, Qh, nullptr,
                                         MROWS, D_, D_, 1);
  gemm_wmma_bf16<<<gg, 256, 0, stream>>>(xb, WkT, nullptr, Kh, nullptr,
                                         MROWS, D_, D_, 1);
  gemm_wmma_bf16<<<gg, 256, 0, stream>>>(xb, WvT, nullptr, Vt, nullptr,
                                         MROWS, D_, D_, 2);

  // 3) causal flash attention: B*H*(T/16) waves, 8 waves per block
  attn_wmma<<<(B_ * H_ * (T_ / 16)) / 8, 256, 0, stream>>>(Qh, Kh, Vt, ctx);

  // 4) output projection + bias -> f32
  gemm_wmma_bf16<<<gg, 256, 0, stream>>>(ctx, WoT, out, nullptr, bo,
                                         MROWS, D_, D_, 0);
}
